// SparseMatmul_7284264534723
// MI455X (gfx1250) — compile-verified, matrix operand order checked by reading, not run on hardware
//
#include <hip/hip_runtime.h>

// ---------------------------------------------------------------------------
// Batched GEMM C[B,M,N] = A[B,M,K] @ B[B,K,N], fp32 in/out, bf16x3 split on
// V_WMMA_F32_16X16X32_BF16 (CDNA5 / gfx1250, wave32).
//  - 128x128 C-tile per 256-thread (8-wave) workgroup; 32x64 per wave.
//  - K consumed in 32-wide chunks; LDS double-buffered; global loads for the
//    next chunk are issued before the WMMA block so HBM latency hides under
//    matrix math.
//  - Dims are constexpr (2048^3) so all strided load offsets become 24-bit
//    instruction immediates: near-zero address VALU in the steady state.
// ---------------------------------------------------------------------------

typedef __bf16 bf16_t;
typedef __attribute__((ext_vector_type(16))) __bf16 v16bf;
typedef __attribute__((ext_vector_type(4)))  __bf16 bf4v;
typedef __attribute__((ext_vector_type(8)))  float  v8f;
typedef __attribute__((ext_vector_type(4)))  float  f4v;

constexpr int Mdim = 2048, Ndim = 2048, Kdim = 2048;
constexpr int BM = 128, BN = 128, BK = 32;
constexpr int LDK = 32;                 // LDS row stride (elements)

__device__ __forceinline__ v8f wmma_bf16(v16bf a, v16bf b, v8f c) {
    return __builtin_amdgcn_wmma_f32_16x16x32_bf16(false, a, false, b,
                                                   (short)0, c, false, false);
}

// ISA 7.12.2 16-bit A-matrix 16x32 lane layout (B^T uses the same layout:
// B staged as [n][k], one column n per lane):
//   lanes 0-15 : elems 0-7 = K0..7,  elems 8-15 = K16..23
//   lanes 16-31: elems 0-7 = K8..15, elems 8-15 = K24..31
__device__ __forceinline__ v16bf load_frag(const bf16_t* rowbase, int lo16) {
    union { v16bf v; uint4 q[2]; } u;
    u.q[0] = *(const uint4*)(rowbase + lo16);
    u.q[1] = *(const uint4*)(rowbase + 16 + lo16);
    return u.v;
}

__device__ __forceinline__ void split4(f4v x, bf4v& h, bf4v& l) {
    h = __builtin_convertvector(x, bf4v);       // round-to-nearest bf16
    f4v hf = __builtin_convertvector(h, f4v);   // exact
    l = __builtin_convertvector(x - hf, bf4v);  // residual
}

__global__ __launch_bounds__(256)
void SparseMatmul_7284264534723_kernel(const float* __restrict__ A,
                                       const float* __restrict__ Bm,
                                       float* __restrict__ C) {
    // Double-buffered staging tiles: A row-major [m][k], B transposed [n][k].
    __shared__ alignas(16) bf16_t sAhi[2][BM * LDK];   // 16 KB
    __shared__ alignas(16) bf16_t sAlo[2][BM * LDK];   // 16 KB
    __shared__ alignas(16) bf16_t sBhi[2][BN * LDK];   // 16 KB
    __shared__ alignas(16) bf16_t sBlo[2][BN * LDK];   // 16 KB  (total 64 KB)

    const int tid   = threadIdx.x;
    const int lane  = tid & 31;
    const int wave  = tid >> 5;            // 0..7
    const int mbase = (wave & 3) * 32;     // 4 waves along M
    const int nbase = (wave >> 2) * 64;    // 2 waves along N, 64 cols each
    const int l15   = lane & 15;
    const int lo16  = (lane < 16) ? 0 : 8;
    const int rhalf = (lane >> 4) * 8;

    const int batch = blockIdx.z;
    const int m0    = blockIdx.y * BM;
    const int n0    = blockIdx.x * BN;

    const float* Ab = A  + (size_t)batch * Mdim * Kdim;
    const float* Bb = Bm + (size_t)batch * Kdim * Ndim;
    float*       Cb = C  + (size_t)batch * Mdim * Ndim;

    // Staging map. A: thread -> (row 0..127, 16-float half). 16 floats/thread.
    const int arow = tid >> 1;
    const int acol = (tid & 1) * 16;
    // B: thread -> column n (0..127), k-groups {bkg+8i .. +3}. 16 floats/thread.
    const int bn   = tid & 127;
    const int bkg  = (tid >> 7) * 4;       // 0 or 4

    const float* aPtr = Ab + (size_t)(m0 + arow) * Kdim + acol;
    const float* bPtr = Bb + (size_t)bkg * Ndim + n0 + bn;

    f4v aReg[4];
    f4v bReg[4];

    auto issue_loads = [&]() {
#pragma unroll
        for (int j = 0; j < 4; ++j)
            aReg[j] = *(const f4v*)(aPtr + j * 4);
#pragma unroll
        for (int i = 0; i < 4; ++i)
#pragma unroll
            for (int j = 0; j < 4; ++j)
                bReg[i][j] = bPtr[(size_t)(i * 8 + j) * Ndim];  // imm offsets
        aPtr += BK;
        bPtr += (size_t)BK * Ndim;
    };

    auto convert_store = [&](int buf) {
#pragma unroll
        for (int j = 0; j < 4; ++j) {
            bf4v h, l;
            split4(aReg[j], h, l);
            *(bf4v*)&sAhi[buf][arow * LDK + acol + j * 4] = h;
            *(bf4v*)&sAlo[buf][arow * LDK + acol + j * 4] = l;
        }
#pragma unroll
        for (int i = 0; i < 4; ++i) {
            bf4v h, l;
            split4(bReg[i], h, l);
            *(bf4v*)&sBhi[buf][bn * LDK + bkg + i * 8] = h;
            *(bf4v*)&sBlo[buf][bn * LDK + bkg + i * 8] = l;
        }
    };

    v8f acc[2][4];
#pragma unroll
    for (int i = 0; i < 2; ++i)
#pragma unroll
        for (int j = 0; j < 4; ++j)
#pragma unroll
            for (int g = 0; g < 8; ++g) acc[i][j][g] = 0.0f;

    // Prologue: stage chunk 0.
    issue_loads();
    convert_store(0);

    int p = 0;
    for (int kt = 0; kt < Kdim; kt += BK) {
        __syncthreads();                       // buf[p] ready; buf[p^1] free
        const bool more = (kt + BK) < Kdim;
        if (more) issue_loads();               // HBM loads in flight under WMMA

        // ---- A fragments for this wave's two m-tiles
        v16bf ahi[2], alo[2];
#pragma unroll
        for (int mt = 0; mt < 2; ++mt) {
            const int r = mbase + mt * 16 + l15;
            ahi[mt] = load_frag(&sAhi[p][r * LDK], lo16);
            alo[mt] = load_frag(&sAlo[p][r * LDK], lo16);
        }
        // ---- 4 n-tiles in two halves (caps live B fragments at 32 VGPRs)
#pragma unroll
        for (int half = 0; half < 2; ++half) {
            v16bf bhi[2], blo[2];
#pragma unroll
            for (int t = 0; t < 2; ++t) {
                const int c = nbase + (half * 2 + t) * 16 + l15;
                bhi[t] = load_frag(&sBhi[p][c * LDK], lo16);
                blo[t] = load_frag(&sBlo[p][c * LDK], lo16);
            }
#pragma unroll
            for (int mt = 0; mt < 2; ++mt)
#pragma unroll
                for (int t = 0; t < 2; ++t) {
                    const int nt = half * 2 + t;
                    // C += Ahi*Bhi + Ahi*Blo + Alo*Bhi
                    acc[mt][nt] = wmma_bf16(ahi[mt], bhi[t], acc[mt][nt]);
                    acc[mt][nt] = wmma_bf16(ahi[mt], blo[t], acc[mt][nt]);
                    acc[mt][nt] = wmma_bf16(alo[mt], bhi[t], acc[mt][nt]);
                }
        }

        if (more) convert_store(p ^ 1);        // wait loads AFTER the WMMAs
        p ^= 1;
    }

    // ---- Epilogue: VGPR g -> M = g (lanes 0-15) / g+8 (lanes 16-31)
#pragma unroll
    for (int mt = 0; mt < 2; ++mt) {
#pragma unroll
        for (int nt = 0; nt < 4; ++nt) {
            const int rowbase = m0 + mbase + mt * 16 + rhalf;
            const int col     = n0 + nbase + nt * 16 + l15;
#pragma unroll
            for (int g = 0; g < 8; ++g) {
                Cb[(size_t)(rowbase + g) * Ndim + col] = acc[mt][nt][g];
            }
        }
    }
}

extern "C" void kernel_launch(void* const* d_in, const int* in_sizes, int n_in,
                              void* d_out, int out_size, void* d_ws, size_t ws_size,
                              hipStream_t stream) {
    const float* A  = (const float*)d_in[0];
    const float* Bm = (const float*)d_in[1];
    float*       C  = (float*)d_out;

    const int batches = out_size / (Mdim * Ndim);     // = 4
    dim3 grid(Ndim / BN, Mdim / BM, batches);         // (16, 16, 4)
    dim3 block(256);
    SparseMatmul_7284264534723_kernel<<<grid, block, 0, stream>>>(A, Bm, C);
}
